// MultiHeadAttention_70600672411714
// MI455X (gfx1250) — matrix operand order repaired and checked by reading, not compiled, not run on hardware
//
#include <hip/hip_runtime.h>

typedef __bf16 bf16_t;
typedef __attribute__((ext_vector_type(16))) __bf16 v16bf;
typedef __attribute__((ext_vector_type(8)))  __bf16 v8bf;
typedef __attribute__((ext_vector_type(8)))  float  v8f;

#define HIDDEN   1024
#define HEADS    16
#define HEAD_DIM 64
#define SEQ      2048
#define BATCH    2
#define MROWS    (BATCH * SEQ)   /* 4096 */

// ---------------------------------------------------------------------------
// WMMA helper: D = A(16x32 bf16) * B(32x16 bf16) + C(16x16 f32)
// ---------------------------------------------------------------------------
__device__ __forceinline__ v8f wmma_bf16(v16bf a, v16bf b, v8f c) {
  return __builtin_amdgcn_wmma_f32_16x16x32_bf16(
      /*neg_a=*/false, a, /*neg_b=*/false, b,
      /*c_mod=*/(short)0, c, /*reuse_a=*/false, /*reuse_b=*/false);
}

// A-fragment loader (contiguous K order, matching B):
// lane<16 : row = lane,    K = {kb..kb+15}
// lane>=16: row = lane-16, K = {kb+16..kb+31}; caller passes k0 = kb + 8*khalf.
__device__ __forceinline__ v16bf ld_a_frag(const bf16_t* rowPtr, int k0) {
  const int kc = (k0 & ~15) + 2 * (k0 & 15);
  v8bf lo = *reinterpret_cast<const v8bf*>(rowPtr + kc);
  v8bf hi = *reinterpret_cast<const v8bf*>(rowPtr + kc + 8);
  v16bf r;
#pragma unroll
  for (int i = 0; i < 8; ++i) { r[i] = lo[i]; r[i + 8] = hi[i]; }
  return r;
}

// CDNA5 async global->LDS copies (ASYNCcnt-tracked) + wait
__device__ __forceinline__ void async_copy_b64(uint32_t lds_addr, const void* gaddr) {
  asm volatile("global_load_async_to_lds_b64 %0, %1, off"
               :: "v"(lds_addr), "v"((unsigned long long)(uintptr_t)gaddr)
               : "memory");
}
__device__ __forceinline__ void async_copy_b128(uint32_t lds_addr, const void* gaddr) {
  asm volatile("global_load_async_to_lds_b128 %0, %1, off"
               :: "v"(lds_addr), "v"((unsigned long long)(uintptr_t)gaddr)
               : "memory");
}
__device__ __forceinline__ void wait_asynccnt0() {
  asm volatile("s_wait_asynccnt 0x0" ::: "memory");
}

// ---------------------------------------------------------------------------
// Elementwise converts (bandwidth-trivial; everything then lives in L2)
// ---------------------------------------------------------------------------
__global__ void cvt_bf16(const float* __restrict__ in, bf16_t* __restrict__ out, int n) {
  for (int i = blockIdx.x * blockDim.x + threadIdx.x; i < n; i += gridDim.x * blockDim.x)
    out[i] = (bf16_t)in[i];
}

// W [Kd][Nd] row-major fp32 -> WT [Nd][Kd] bf16 (B-fragment friendly layout)
__global__ void cvt_transpose(const float* __restrict__ W, bf16_t* __restrict__ WT,
                              int Kd, int Nd) {
  int n = Kd * Nd;
  for (int i = blockIdx.x * blockDim.x + threadIdx.x; i < n; i += gridDim.x * blockDim.x) {
    int kr = i / Nd, nc = i % Nd;
    WT[(size_t)nc * Kd + kr] = (bf16_t)W[i];
  }
}

// ---------------------------------------------------------------------------
// WMMA GEMM: C[M,N] = A[M,K](bf16) * BT[N,K]^T(bf16) + bias
// 256 thr = 8 waves, wave = 16x64 tile. A tile (32x32 per K-step) staged in LDS
// via double-buffered global_load_async_to_lds_b64 overlapped with WMMAs;
// K-loop unrolled x2 over the two buffers.
// MODE 0: fp32 out.  MODE 1: scatter q/k/v [B,H,S,Hd] bf16, q pre-scaled.
// ---------------------------------------------------------------------------
template <int MODE>
__global__ __launch_bounds__(256) void gemm_wmma(
    const bf16_t* __restrict__ A, const bf16_t* __restrict__ BT,
    const float* __restrict__ bias, int M, int N, int K,
    float* __restrict__ outF, bf16_t* __restrict__ qb,
    bf16_t* __restrict__ kbuf, bf16_t* __restrict__ vbuf) {
  (void)M;
  __shared__ __align__(16) bf16_t As[2][32 * 32];   // double-buffered 32x32 tile

  const int tid   = threadIdx.x;
  const int lane  = tid & 31;
  const int wave  = tid >> 5;
  const int khalf = lane >> 4;
  const int nlo   = lane & 15;
  const int row0  = blockIdx.x * 32 + (wave >> 2) * 16;
  const int col0  = blockIdx.y * 256 + (wave & 3) * 64;
  const int rowB  = blockIdx.x * 32;

  // Per-thread staging slot: 8 bytes (4 bf16) of the 32x32 tile
  const int srow = tid >> 3;                 // 0..31
  const int scol = (tid & 7) * 4;            // 0,4,..,28
  const bf16_t* grow = A + (size_t)(rowB + srow) * K + scol;
  const uint32_t lds0 = (uint32_t)(uintptr_t)&As[0][srow * 32 + scol];
  const uint32_t lds1 = (uint32_t)(uintptr_t)&As[1][srow * 32 + scol];

  v8f acc[4] = {};

  async_copy_b64(lds0, grow);                // stage tile 0
  wait_asynccnt0();
  __syncthreads();

  const bf16_t* alds = &As[0][((wave >> 2) * 16 + nlo) * 32];

  // K = 1024 -> 32 steps of 32, unrolled x2 over the double buffer
  for (int kk = 0; kk < K; kk += 64) {
    // ---- buffer 0 holds [kk, kk+32); stage [kk+32, kk+64) into buffer 1 ----
    async_copy_b64(lds1, grow + kk + 32);
    {
      v16bf a = ld_a_frag(alds, 8 * khalf);
#pragma unroll
      for (int t = 0; t < 4; ++t) {
        const bf16_t* bp = BT + (size_t)(col0 + t * 16 + nlo) * K + kk + 16 * khalf;
        __builtin_prefetch(bp + 32, 0, 3);
        v16bf b = *reinterpret_cast<const v16bf*>(bp);
        acc[t] = wmma_bf16(a, b, acc[t]);
      }
    }
    wait_asynccnt0();
    __syncthreads();

    // ---- buffer 1 holds [kk+32, kk+64); stage [kk+64, kk+96) into buffer 0 ----
    if (kk + 64 < K) async_copy_b64(lds0, grow + kk + 64);
    {
      v16bf a = ld_a_frag(alds + 32 * 32, 8 * khalf);
#pragma unroll
      for (int t = 0; t < 4; ++t) {
        const bf16_t* bp = BT + (size_t)(col0 + t * 16 + nlo) * K + (kk + 32) + 16 * khalf;
        __builtin_prefetch(bp + 32, 0, 3);
        v16bf b = *reinterpret_cast<const v16bf*>(bp);
        acc[t] = wmma_bf16(a, b, acc[t]);
      }
    }
    wait_asynccnt0();
    __syncthreads();
  }

#pragma unroll
  for (int t = 0; t < 4; ++t) {
    const int c  = col0 + t * 16 + nlo;
    const float bv = bias[c];
#pragma unroll
    for (int j = 0; j < 8; ++j) {
      const int r = row0 + j + 8 * khalf;      // C/D layout: row = j + 8*(lane>=16)
      const float val = acc[t][j] + bv;
      if (MODE == 0) {
        outF[(size_t)r * N + c] = val;
      } else {
        const int part = c >> 10, cc = c & 1023;
        const int head = cc >> 6, hd = cc & 63;
        const int bi = r >> 11, s = r & (SEQ - 1);
        const size_t idx = (((size_t)(bi * HEADS + head)) * SEQ + s) * HEAD_DIM + hd;
        if (part == 0)      qb[idx]   = (bf16_t)(val * 0.125f);  // 1/sqrt(64)
        else if (part == 1) kbuf[idx] = (bf16_t)val;
        else                vbuf[idx] = (bf16_t)val;
      }
    }
  }
}

// ---------------------------------------------------------------------------
// Online-softmax update for one 16x32 score block.
// MASK=true only on the causal diagonal band.
// ---------------------------------------------------------------------------
template <bool MASK>
__device__ __forceinline__ void softmax_block(
    const v8f& s0, const v8f& s1, int qr0, int kk0, int khalf, int nlo,
    float* rmax, float* rsum, float* corr, bf16_t* Pw) {
#pragma unroll
  for (int j = 0; j < 8; ++j) {
    float a, b;
    if (MASK) {
      const int qrow_g = qr0 + j + 8 * khalf;
      const int c0 = kk0 + nlo;
      a = (c0 <= qrow_g)      ? s0[j] : -3.0e38f;
      b = (c0 + 16 <= qrow_g) ? s1[j] : -3.0e38f;
    } else {
      a = s0[j];
      b = s1[j];
    }
    float mx = fmaxf(a, b);
    mx = fmaxf(mx, __shfl_xor(mx, 1));
    mx = fmaxf(mx, __shfl_xor(mx, 2));
    mx = fmaxf(mx, __shfl_xor(mx, 4));
    mx = fmaxf(mx, __shfl_xor(mx, 8));
    const float nm = fmaxf(rmax[j], mx);
    const float cj = __expf(rmax[j] - nm);
    const float pa = __expf(a - nm);
    const float pb = __expf(b - nm);
    float rs = pa + pb;
    rs += __shfl_xor(rs, 1);
    rs += __shfl_xor(rs, 2);
    rs += __shfl_xor(rs, 4);
    rs += __shfl_xor(rs, 8);
    rsum[j] = rsum[j] * cj + rs;
    rmax[j] = nm;
    corr[j] = cj;
    const int m = j + 8 * khalf;
    Pw[m * 32 + nlo]      = (bf16_t)pa;
    Pw[m * 32 + 16 + nlo] = (bf16_t)pb;
  }
}

// ---------------------------------------------------------------------------
// Causal flash attention. Grid: B*H*16 blocks; block = 256 thr = 8 waves.
// Block owns (b,h) and a 128-row q segment; wave owns 16 q rows.
// K block async-staged to LDS (shared by all 8 waves); V block staged
// transposed with one coalesced 16B load + 8 ds_store_b16 per thread.
// ---------------------------------------------------------------------------
__global__ __launch_bounds__(256) void flash_attn(
    const bf16_t* __restrict__ q, const bf16_t* __restrict__ k,
    const bf16_t* __restrict__ v, bf16_t* __restrict__ ctx) {
  __shared__ __align__(16) bf16_t Ks[32 * HEAD_DIM];  // K rows [key][hd]  (4 KB)
  __shared__ __align__(16) bf16_t VT[HEAD_DIM * 32];  // V^T    [hd][key]  (4 KB)
  __shared__ __align__(16) bf16_t Ps[8 * 16 * 32];    // per-wave P 16x32  (8 KB)

  const int tid   = threadIdx.x;
  const int lane  = tid & 31;
  const int wave  = tid >> 5;
  const int khalf = lane >> 4;
  const int nlo   = lane & 15;
  const int bh    = blockIdx.x >> 4;     // 0..31
  const int qseg  = blockIdx.x & 15;
  const int bi    = bh >> 4;
  const int head  = bh & 15;
  const int qr0   = qseg * 128 + wave * 16;

  const bf16_t* Qb = q + (size_t)bh * SEQ * HEAD_DIM;
  const bf16_t* Kb = k + (size_t)bh * SEQ * HEAD_DIM;
  const bf16_t* Vb = v + (size_t)bh * SEQ * HEAD_DIM;

  // Persistent Q fragments (two Hd chunks of 32)
  const bf16_t* qrow = Qb + (size_t)(qr0 + nlo) * HEAD_DIM;
  v16bf aq0 = ld_a_frag(qrow, 0 + 8 * khalf);
  v16bf aq1 = ld_a_frag(qrow, 32 + 8 * khalf);

  v8f acc[4] = {};
  float rmax[8], rsum[8], corr[8];
#pragma unroll
  for (int j = 0; j < 8; ++j) { rmax[j] = -1.0e30f; rsum[j] = 0.0f; }

  bf16_t* Pw = Ps + wave * (16 * 32);

  // Per-thread staging slots: row/key = tid>>3 (0..31), 8-elem column chunk
  const int skey = tid >> 3;
  const int shd  = (tid & 7) * 8;
  const uint32_t ldsK = (uint32_t)(uintptr_t)&Ks[skey * HEAD_DIM + shd];

  const int nkb = (qseg + 1) * 4;        // causal: keys up to qseg*128+127

  for (int kbk = 0; kbk < nkb; ++kbk) {
    const int kk0 = kbk * 32;

    __syncthreads();                     // all waves done with previous tiles

    // Async-stage K block (16B per thread, ASYNCcnt-tracked)
    async_copy_b128(ldsK, Kb + (size_t)(kk0 + skey) * HEAD_DIM + shd);

    // Stage V^T: one coalesced 16B load, 8 scattered 2B LDS stores
    {
      v8bf vv = *reinterpret_cast<const v8bf*>(
          Vb + (size_t)(kk0 + skey) * HEAD_DIM + shd);
#pragma unroll
      for (int e = 0; e < 8; ++e) VT[(shd + e) * 32 + skey] = vv[e];
    }
    // Warm L2 for next block (global_prefetch_b8)
    if (kbk + 1 < nkb) {
      const size_t nrow = (size_t)(kk0 + 32 + skey) * HEAD_DIM + shd;
      __builtin_prefetch(Vb + nrow, 0, 3);
      __builtin_prefetch(Kb + nrow, 0, 3);
    }
    wait_asynccnt0();
    __syncthreads();

    // Waves whose q rows all precede this key block contribute nothing.
    if (kk0 <= qr0 + 15) {
      // ---- scores: S = Q(16x64) * K^T(64x32) from LDS K tile ----
      v8f s0 = {}, s1 = {};
      {
        const bf16_t* kr0 = Ks + nlo * HEAD_DIM + 16 * khalf;
        const bf16_t* kr1 = Ks + (16 + nlo) * HEAD_DIM + 16 * khalf;
        v16bf b00 = *reinterpret_cast<const v16bf*>(kr0);
        v16bf b10 = *reinterpret_cast<const v16bf*>(kr1);
        v16bf b01 = *reinterpret_cast<const v16bf*>(kr0 + 32);
        v16bf b11 = *reinterpret_cast<const v16bf*>(kr1 + 32);
        s0 = wmma_bf16(aq0, b00, s0);
        s1 = wmma_bf16(aq0, b10, s1);
        s0 = wmma_bf16(aq1, b01, s0);
        s1 = wmma_bf16(aq1, b11, s1);
      }

      // Mask only on the diagonal band (wave-uniform branch)
      if (kk0 + 31 > qr0)
        softmax_block<true >(s0, s1, qr0, kk0, khalf, nlo, rmax, rsum, corr, Pw);
      else
        softmax_block<false>(s0, s1, qr0, kk0, khalf, nlo, rmax, rsum, corr, Pw);

#pragma unroll
      for (int t = 0; t < 4; ++t)
#pragma unroll
        for (int j = 0; j < 8; ++j) acc[t][j] *= corr[j];

      // ---- context: acc += P(16x32) * V(32x64) ----
      v16bf ap = ld_a_frag(Pw + nlo * 32, 8 * khalf);
#pragma unroll
      for (int t = 0; t < 4; ++t) {
        v16bf bv = *reinterpret_cast<const v16bf*>(VT + (t * 16 + nlo) * 32 + 16 * khalf);
        acc[t] = wmma_bf16(ap, bv, acc[t]);
      }
    }
  }

  // ---- normalize + store ctx as bf16 [B*S, D] ----
#pragma unroll
  for (int t = 0; t < 4; ++t)
#pragma unroll
    for (int j = 0; j < 8; ++j) {
      const int m = j + 8 * khalf;
      const int srow = qr0 + m;
      const int col = head * HEAD_DIM + t * 16 + nlo;
      const float val = acc[t][j] / rsum[j];
      ctx[((size_t)(bi * SEQ + srow)) * HIDDEN + col] = (bf16_t)val;
    }
}

// ---------------------------------------------------------------------------
// Host launch
// ---------------------------------------------------------------------------
extern "C" void kernel_launch(void* const* d_in, const int* in_sizes, int n_in,
                              void* d_out, int out_size, void* d_ws, size_t ws_size,
                              hipStream_t stream) {
  (void)in_sizes; (void)n_in; (void)out_size; (void)ws_size;
  const float* x      = (const float*)d_in[0];
  const float* W_attn = (const float*)d_in[1];
  const float* b_attn = (const float*)d_in[2];
  const float* W_proj = (const float*)d_in[3];
  const float* b_proj = (const float*)d_in[4];
  float* out = (float*)d_out;

  char* ws = (char*)d_ws;                       // 48 MiB total
  bf16_t* xb   = (bf16_t*)(ws);                 //  8 MiB  x bf16 [4096][1024]
  bf16_t* WaT  = (bf16_t*)(ws + (8u  << 20));   //  6 MiB  W_attn^T [3072][1024]
  bf16_t* WpT  = (bf16_t*)(ws + (14u << 20));   //  2 MiB  W_proj^T [1024][1024]
  bf16_t* qb   = (bf16_t*)(ws + (16u << 20));   //  8 MiB  q [B,H,S,64] (pre-scaled)
  bf16_t* kb   = (bf16_t*)(ws + (24u << 20));   //  8 MiB
  bf16_t* vb   = (bf16_t*)(ws + (32u << 20));   //  8 MiB
  bf16_t* ctxb = (bf16_t*)(ws + (40u << 20));   //  8 MiB  ctx bf16 [4096][1024]

  cvt_bf16<<<1024, 256, 0, stream>>>(x, xb, MROWS * HIDDEN);
  cvt_transpose<<<1024, 256, 0, stream>>>(W_attn, WaT, HIDDEN, 3 * HIDDEN);
  cvt_transpose<<<1024, 256, 0, stream>>>(W_proj, WpT, HIDDEN, HIDDEN);

  dim3 g1(MROWS / 32, (3 * HIDDEN) / 256);
  gemm_wmma<1><<<g1, 256, 0, stream>>>(xb, WaT, b_attn, MROWS, 3 * HIDDEN, HIDDEN,
                                       nullptr, qb, kb, vb);

  flash_attn<<<BATCH * HEADS * 16, 256, 0, stream>>>(qb, kb, vb, ctxb);

  dim3 g2(MROWS / 32, HIDDEN / 256);
  gemm_wmma<0><<<g2, 256, 0, stream>>>(ctxb, WpT, b_proj, MROWS, HIDDEN, HIDDEN,
                                       out, nullptr, nullptr, nullptr);
}
